// SelectiveRelOnlyEncodeDecodeMLP_68599217651898
// MI455X (gfx1250) — matrix operand order repaired and checked by reading, compiled, not run on hardware
//
#include <hip/hip_runtime.h>
#include <hip/hip_bf16.h>

typedef __attribute__((ext_vector_type(16))) _Float16 v16h;
typedef __attribute__((ext_vector_type(8)))  _Float16 v8h;
typedef __attribute__((ext_vector_type(8)))  float    v8f;

#define HID 128
#define FRAG_H 512              // halfs per 32x16 (or 16x32) f16 fragment
#define W1F_OFF 0               // 8 fragments  (16->128, K padded to 32)
#define W2F_OFF (8 * FRAG_H)    // 32 fragments (128->128)
#define W3F_OFF (40 * FRAG_H)   // 32 fragments (128->128)
#define W4F_OFF (72 * FRAG_H)   // 4 fragments  (128->16)
#define N_FRAGS 76

// ---------------------------------------------------------------------------
// Convert fp32 weights into f16 WMMA B-fragments (contiguous-K B order used here, matching the A loads:
// lane = {khalf = lane/16, n = lane%16}; element j -> K = kt*32 + khalf*16 + j)
// Each fragment is stored as frag[lane*16 + j] so a lane's slice is a
// contiguous 32-byte load.
// ---------------------------------------------------------------------------
__global__ void prep_weights_kernel(const float* __restrict__ w1,
                                    const float* __restrict__ w2,
                                    const float* __restrict__ w3,
                                    const float* __restrict__ w4,
                                    _Float16* __restrict__ wsf) {
    int f = blockIdx.x;
    int e = threadIdx.x;            // 0..511
    int lane = e >> 4;
    int j    = e & 15;
    int nloc  = lane & 15;
    int khalf = lane >> 4;

    const float* W;
    int Kdim, Ncols, kt, t, base;
    if (f < 8) {
        W = w1; Kdim = 16; Ncols = HID; kt = 0; t = f;
        base = W1F_OFF + f * FRAG_H;
    } else if (f < 40) {
        int g = f - 8;
        W = w2; Kdim = HID; Ncols = HID; kt = g & 3; t = g >> 2;
        base = W2F_OFF + g * FRAG_H;
    } else if (f < 72) {
        int g = f - 40;
        W = w3; Kdim = HID; Ncols = HID; kt = g & 3; t = g >> 2;
        base = W3F_OFF + g * FRAG_H;
    } else {
        int g = f - 72;
        W = w4; Kdim = HID; Ncols = 16; kt = g; t = 0;
        base = W4F_OFF + g * FRAG_H;
    }

    int k = kt * 32 + khalf * 16 + j;
    int n = t * 16 + nloc;
    float v = (k < Kdim) ? W[k * Ncols + n] : 0.0f;
    wsf[base + lane * 16 + j] = (_Float16)v;
}

// ---------------------------------------------------------------------------
// Main kernel: one 16-edge tile per wave, full 4-layer MLP via WMMA.
// ---------------------------------------------------------------------------
__launch_bounds__(256)
__global__ void pose_mlp_kernel(const float* __restrict__ edges,
                                const _Float16* __restrict__ wsf,
                                const float* __restrict__ b1,
                                const float* __restrict__ b2,
                                const float* __restrict__ b3,
                                const float* __restrict__ b4,
                                float* __restrict__ out,
                                int n) {
    // per-wave LDS: 256 halfs rel-pose staging + 16x128 halfs activations
    __shared__ __align__(16) _Float16 lds[8 * (256 + 2048)];

    const int lane  = threadIdx.x & 31;
    const int wave  = threadIdx.x >> 5;
    const int tile  = blockIdx.x * 8 + wave;
    const int ntile = (n + 15) >> 4;
    if (tile >= ntile) return;

    _Float16* rp = lds + wave * (256 + 2048);
    _Float16* Hb = rp + 256;

    const int m     = lane & 15;   // A-matrix row / D-matrix column index
    const int khalf = lane >> 4;
    const int base_e = tile * 16;

    // ---- rel-pose computation (fp32), lanes 0..15, one edge each ----
    if (lane < 16) {
        int e = base_e + m;
        if (e < n) {
            const float2* ep = (const float2*)(edges + (long)e * 14);
            float2 p0 = ep[0], p1 = ep[1], p2 = ep[2], p3 = ep[3];
            float2 p4 = ep[4], p5 = ep[5], p6 = ep[6];
            float qw = p0.x, qx = p0.y, qy = p1.x, qz = p1.y;
            float t1x = p2.x, t1y = p2.y, t1z = p3.x;
            float pw = p3.y, px = p4.x, py = p4.y, pz = p5.x;
            float t2x = p5.y, t2y = p6.x, t2z = p6.y;

            float a00 = 1.f - 2.f*(qy*qy + qz*qz), a01 = 2.f*(qx*qy - qz*qw), a02 = 2.f*(qx*qz + qy*qw);
            float a10 = 2.f*(qx*qy + qz*qw), a11 = 1.f - 2.f*(qx*qx + qz*qz), a12 = 2.f*(qy*qz - qx*qw);
            float a20 = 2.f*(qx*qz - qy*qw), a21 = 2.f*(qy*qz + qx*qw), a22 = 1.f - 2.f*(qx*qx + qy*qy);

            float c00 = 1.f - 2.f*(py*py + pz*pz), c01 = 2.f*(px*py - pz*pw), c02 = 2.f*(px*pz + py*pw);
            float c10 = 2.f*(px*py + pz*pw), c11 = 1.f - 2.f*(px*px + pz*pz), c12 = 2.f*(py*pz - px*pw);
            float c20 = 2.f*(px*pz - py*pw), c21 = 2.f*(py*pz + px*pw), c22 = 1.f - 2.f*(px*px + py*py);

            // R_rel = R1 * R2^T
            float r00 = a00*c00 + a01*c01 + a02*c02;
            float r01 = a00*c10 + a01*c11 + a02*c12;
            float r02 = a00*c20 + a01*c21 + a02*c22;
            float r10 = a10*c00 + a11*c01 + a12*c02;
            float r11 = a10*c10 + a11*c11 + a12*c12;
            float r12 = a10*c20 + a11*c21 + a12*c22;
            float r20 = a20*c00 + a21*c01 + a22*c02;
            float r21 = a20*c10 + a21*c11 + a22*c12;
            float r22 = a20*c20 + a21*c21 + a22*c22;

            float tx = t1x - (r00*t2x + r01*t2y + r02*t2z);
            float ty = t1y - (r10*t2x + r11*t2y + r12*t2z);
            float tz = t1z - (r20*t2x + r21*t2y + r22*t2z);

            union { _Float16 h[16]; uint4 v[2]; } u;
            u.h[0] = (_Float16)r00; u.h[1] = (_Float16)r01; u.h[2] = (_Float16)r02; u.h[3]  = (_Float16)tx;
            u.h[4] = (_Float16)r10; u.h[5] = (_Float16)r11; u.h[6] = (_Float16)r12; u.h[7]  = (_Float16)ty;
            u.h[8] = (_Float16)r20; u.h[9] = (_Float16)r21; u.h[10] = (_Float16)r22; u.h[11] = (_Float16)tz;
            u.h[12] = (_Float16)0.f; u.h[13] = (_Float16)0.f; u.h[14] = (_Float16)0.f; u.h[15] = (_Float16)1.f;

            uint4* rpv = (uint4*)rp;
            rpv[m * 2 + 0] = u.v[0];
            rpv[m * 2 + 1] = u.v[1];
        }
    }
    __builtin_amdgcn_wave_barrier();   // keep ds store->load order across reconvergence

    // ---- Layer 1: A = rel_pose (16x16, K zero-padded to 32) ----
    v16h a1;
    {
        v8h lo = *(const v8h*)(rp + m * 16);
        v8h hi = *(const v8h*)(rp + m * 16 + 8);
#pragma unroll
        for (int i = 0; i < 8; ++i) { a1[i] = khalf ? (_Float16)0.f : lo[i]; a1[8 + i] = khalf ? (_Float16)0.f : hi[i]; }
    }
#pragma unroll
    for (int t = 0; t < 8; ++t) {
        const v16h b = *(const v16h*)(wsf + W1F_OFF + t * FRAG_H + lane * 16);
        v8f acc = {};
        acc = __builtin_amdgcn_wmma_f32_16x16x32_f16(false, a1, false, b, (short)0, acc, false, false);
        float bias = b1[t * 16 + m];
#pragma unroll
        for (int r = 0; r < 8; ++r) {
            float h = acc[r] + bias;
            h = h > 0.f ? h : 0.f;
            Hb[(r + 8 * khalf) * HID + t * 16 + m] = (_Float16)h;
        }
    }
    __builtin_amdgcn_wave_barrier();

    v16h a[4];

    // ---- Layer 2 ----
#pragma unroll
    for (int kt = 0; kt < 4; ++kt) {
        v8h lo = *(const v8h*)(Hb + m * HID + kt * 32 + khalf * 16);
        v8h hi = *(const v8h*)(Hb + m * HID + kt * 32 + khalf * 16 + 8);
#pragma unroll
        for (int i = 0; i < 8; ++i) { a[kt][i] = lo[i]; a[kt][8 + i] = hi[i]; }
    }
    __builtin_amdgcn_wave_barrier();
#pragma unroll
    for (int t = 0; t < 8; ++t) {
        v8f acc = {};
#pragma unroll
        for (int kt = 0; kt < 4; ++kt) {
            const v16h b = *(const v16h*)(wsf + W2F_OFF + (t * 4 + kt) * FRAG_H + lane * 16);
            acc = __builtin_amdgcn_wmma_f32_16x16x32_f16(false, a[kt], false, b, (short)0, acc, false, false);
        }
        float bias = b2[t * 16 + m];
#pragma unroll
        for (int r = 0; r < 8; ++r) {
            float h = acc[r] + bias;
            h = h > 0.f ? h : 0.f;
            Hb[(r + 8 * khalf) * HID + t * 16 + m] = (_Float16)h;
        }
    }
    __builtin_amdgcn_wave_barrier();

    // ---- Layer 3 ----
#pragma unroll
    for (int kt = 0; kt < 4; ++kt) {
        v8h lo = *(const v8h*)(Hb + m * HID + kt * 32 + khalf * 16);
        v8h hi = *(const v8h*)(Hb + m * HID + kt * 32 + khalf * 16 + 8);
#pragma unroll
        for (int i = 0; i < 8; ++i) { a[kt][i] = lo[i]; a[kt][8 + i] = hi[i]; }
    }
    __builtin_amdgcn_wave_barrier();
#pragma unroll
    for (int t = 0; t < 8; ++t) {
        v8f acc = {};
#pragma unroll
        for (int kt = 0; kt < 4; ++kt) {
            const v16h b = *(const v16h*)(wsf + W3F_OFF + (t * 4 + kt) * FRAG_H + lane * 16);
            acc = __builtin_amdgcn_wmma_f32_16x16x32_f16(false, a[kt], false, b, (short)0, acc, false, false);
        }
        float bias = b3[t * 16 + m];
#pragma unroll
        for (int r = 0; r < 8; ++r) {
            float h = acc[r] + bias;
            h = h > 0.f ? h : 0.f;
            Hb[(r + 8 * khalf) * HID + t * 16 + m] = (_Float16)h;
        }
    }
    __builtin_amdgcn_wave_barrier();

    // ---- Layer 4: 128 -> 16, no ReLU, write fp32 output ----
#pragma unroll
    for (int kt = 0; kt < 4; ++kt) {
        v8h lo = *(const v8h*)(Hb + m * HID + kt * 32 + khalf * 16);
        v8h hi = *(const v8h*)(Hb + m * HID + kt * 32 + khalf * 16 + 8);
#pragma unroll
        for (int i = 0; i < 8; ++i) { a[kt][i] = lo[i]; a[kt][8 + i] = hi[i]; }
    }
    {
        v8f acc = {};
#pragma unroll
        for (int kt = 0; kt < 4; ++kt) {
            const v16h b = *(const v16h*)(wsf + W4F_OFF + kt * FRAG_H + lane * 16);
            acc = __builtin_amdgcn_wmma_f32_16x16x32_f16(false, a[kt], false, b, (short)0, acc, false, false);
        }
        float bias = b4[m];
#pragma unroll
        for (int r = 0; r < 8; ++r) {
            int edge = base_e + r + 8 * khalf;
            if (edge < n) out[(long)edge * 16 + m] = acc[r] + bias;
        }
    }
}

extern "C" void kernel_launch(void* const* d_in, const int* in_sizes, int n_in,
                              void* d_out, int out_size, void* d_ws, size_t ws_size,
                              hipStream_t stream) {
    const float* edges = (const float*)d_in[0];
    const float* w1 = (const float*)d_in[1];
    const float* b1 = (const float*)d_in[2];
    const float* w2 = (const float*)d_in[3];
    const float* b2 = (const float*)d_in[4];
    const float* w3 = (const float*)d_in[5];
    const float* b3 = (const float*)d_in[6];
    const float* w4 = (const float*)d_in[7];
    const float* b4 = (const float*)d_in[8];
    float* out = (float*)d_out;
    _Float16* wsf = (_Float16*)d_ws;

    int n = in_sizes[0] / 14;

    prep_weights_kernel<<<N_FRAGS, 512, 0, stream>>>(w1, w2, w3, w4, wsf);

    int ntiles = (n + 15) / 16;
    int blocks = (ntiles + 7) / 8;
    pose_mlp_kernel<<<blocks, 256, 0, stream>>>(edges, wsf, b1, b2, b3, b4, out, n);
}
